// Powerful_28123445854579
// MI455X (gfx1250) — matrix operand order repaired and checked by reading, not compiled, not run on hardware
//
#include <hip/hip_runtime.h>
#include <math.h>

#define NN   512
#define NPOS (512*512)
#define HH   64
#define HFD  64
#define NCLS 10
#define NEDGE 16384
#define BN_EPS 1e-5f
#define KSTEPS (NN/32)

typedef __attribute__((ext_vector_type(16))) __bf16 v16bf;
typedef __attribute__((ext_vector_type(8)))  float  v8f;
typedef __attribute__((ext_vector_type(4)))  int    v4i;

union Frag { v16bf v; uint4 q[2]; };

__device__ __forceinline__ unsigned short f2bf(float f) {
    unsigned int x = __float_as_uint(f);
    x += 0x7fffu + ((x >> 16) & 1u);          // round-to-nearest-even
    return (unsigned short)(x >> 16);
}
__device__ __forceinline__ float bf2f(unsigned short h) {
    return __uint_as_float(((unsigned int)h) << 16);
}

// ---- gfx1250 async global->LDS copy (ASYNCcnt path), builtin if present,
// ---- inline-asm ISA mnemonic otherwise.
#if defined(__has_builtin)
#  if __has_builtin(__builtin_amdgcn_global_load_async_to_lds_b128)
#    define ASYNC_CP16(g, l) __builtin_amdgcn_global_load_async_to_lds_b128( \
        (__attribute__((address_space(1))) v4i*)(g),                         \
        (__attribute__((address_space(3))) v4i*)(l), 0, 0)
#  endif
#  if __has_builtin(__builtin_amdgcn_s_wait_asynccnt)
#    define WAITA0() __builtin_amdgcn_s_wait_asynccnt(0)
#    define WAITA1() __builtin_amdgcn_s_wait_asynccnt(1)
#  endif
#endif
#ifndef ASYNC_CP16
__device__ __forceinline__ unsigned __lds_off(const void* p) {
    return (unsigned)(unsigned long long)(const __attribute__((address_space(3))) void*)p;
}
#  define ASYNC_CP16(g, l) asm volatile(                                     \
      "global_load_async_to_lds_b128 %0, %1, off"                            \
      :: "v"(__lds_off(l)), "v"((const void*)(g)) : "memory")
#endif
#ifndef WAITA0
#  define WAITA0() asm volatile("s_wait_asynccnt 0x0" ::: "memory")
#  define WAITA1() asm volatile("s_wait_asynccnt 0x1" ::: "memory")
#endif

// ---------------------------------------------------------------- utilities
__global__ void k_zero_f32(float* p, int n) {
    int i = blockIdx.x * blockDim.x + threadIdx.x;
    if (i < n) p[i] = 0.f;
}
__global__ void k_cvt_bf16(const float* __restrict__ s, unsigned short* __restrict__ d, int n) {
    int i = blockIdx.x * blockDim.x + threadIdx.x;
    if (i < n) d[i] = f2bf(s[i]);
}

// ------------------------------------------------- adjacency scatter (E*32)
__global__ void k_build_A(const float* __restrict__ x, const int* __restrict__ ei,
                          float* __restrict__ A) {
    int idx = blockIdx.x * blockDim.x + threadIdx.x;   // exact grid E*32
    int e = idx >> 5, f = idx & 31;
    int s = ei[e], d = ei[NEDGE + e];
    float val = (f < 16) ? x[s * 16 + f] : x[d * 16 + (f - 16)];
    atomicAdd(&A[((size_t)s * NN + d) * 32 + f], val);
}

// ------------------------------------------------------------------------
// Generic 64-output-channel GEMM over positions (1x1 convs / m4):
//   out[h][p] = bias[h] + sum_k Wb[h][k] * col_p[k],
// column p = concat(srcA[p][0..KA), srcB[p][0..KB)).  One wave per 16
// positions, all 64 output channels.  Weights are async-staged into LDS
// once per block (padded rows -> conflict-free ds_load_b128 A-fragments);
// B columns are register double-buffered with a branch-free clamped
// prefetch so the WMMA chain only waits on short-latency DS loads.
// mode 0: bf16 channel-major [h][p]; 1: bf16 [h][j*NN+i]; 2: f32 [p][h]
// ------------------------------------------------------------------------
__device__ __forceinline__ Frag loadB_col(const unsigned short* __restrict__ srcA, int KA,
                                          const unsigned short* __restrict__ srcB, int KB,
                                          int p, int kb) {
    const unsigned short* bp = (kb < KA) ? (srcA + (size_t)p * KA + kb)
                                         : (srcB + (size_t)p * KB + (kb - KA));
    Frag B; B.q[0] = *(const uint4*)bp; B.q[1] = *(const uint4*)(bp + 8);
    return B;
}

__global__ void k_gemm_pos(const unsigned short* __restrict__ Wb,
                           const float* __restrict__ bias,
                           const unsigned short* __restrict__ srcA, int KA,
                           const unsigned short* __restrict__ srcB, int KB,
                           unsigned short* __restrict__ out16,
                           float* __restrict__ outf, int mode) {
    __shared__ unsigned short Wlds[64 * 136];   // 64 rows, stride K+8 (max 136)
    int tid  = blockIdx.x * blockDim.x + threadIdx.x;
    int wave = tid >> 5;
    int lane = tid & 31;
    int col  = lane & 15;
    int sel  = lane >> 4;
    int p    = wave * 16 + col;
    int K    = KA + KB;
    int ldsStride = K + 8;

    // async-stage 64xK bf16 weights into LDS (16B chunks)
    int chunksPerRow = K >> 3;
    int totalChunks  = 64 * chunksPerRow;
    for (int c = threadIdx.x; c < totalChunks; c += 256) {
        int row = c / chunksPerRow;
        int idx = c - row * chunksPerRow;
        ASYNC_CP16(Wb + (size_t)row * K + idx * 8, &Wlds[row * ldsStride + idx * 8]);
    }
    WAITA0();
    __syncthreads();

    v8f zro = {0.f,0.f,0.f,0.f,0.f,0.f,0.f,0.f};
    v8f acc[4] = {zro, zro, zro, zro};

    Frag Bcur = loadB_col(srcA, KA, srcB, KB, p, sel * 16);
    for (int k0 = 0; k0 < K; k0 += 32) {
        // branch-free prefetch of next step's B column (clamped; unused on last)
        int kb2 = (k0 + 32 < K) ? (k0 + 32 + sel * 16) : (sel * 16);
        Frag Bnext = loadB_col(srcA, KA, srcB, KB, p, kb2);
        Frag Af[4];
#pragma unroll
        for (int ht = 0; ht < 4; ++ht) {
            const unsigned short* ap = &Wlds[(ht * 16 + col) * ldsStride + k0 + sel * 16];
            Af[ht].q[0] = *(const uint4*)ap;
            Af[ht].q[1] = *(const uint4*)(ap + 8);
        }
#pragma unroll
        for (int ht = 0; ht < 4; ++ht)
            acc[ht] = __builtin_amdgcn_wmma_f32_16x16x32_bf16(
                false, Af[ht].v, false, Bcur.v, (short)0, acc[ht], false, false);
        Bcur = Bnext;
    }
#pragma unroll
    for (int ht = 0; ht < 4; ++ht) {
#pragma unroll
        for (int r = 0; r < 8; ++r) {
            int h = ht * 16 + sel * 8 + r;
            float val = acc[ht][r] + bias[h];
            if (mode == 0) {
                out16[(size_t)h * NPOS + p] = f2bf(val);
            } else if (mode == 1) {
                int i = p >> 9, j = p & (NN - 1);
                out16[(size_t)h * NPOS + (size_t)j * NN + i] = f2bf(val);
            } else {
                outf[(size_t)p * HH + h] = val;
            }
        }
    }
}

// ------------------------------------------------------------------------
// 64 batched 512x512x512 GEMMs: mult[h] = O1[h] @ O2[h].
// Block (8 waves) -> 128x64 C tile of one channel.  The 32x64 B tile is
// shared by all 8 waves: staged via async global->LDS copies, double
// buffered (10 KB, padded 80B rows -> conflict-free ds_load_b128 reads).
// A fragments are per-wave and register double-buffered.
// ------------------------------------------------------------------------
#define BROW 40                       // LDS row stride in halves (80 B)
#define BBUF (64 * BROW)              // one buffer: 64 rows

__global__ void k_mult(const unsigned short* __restrict__ O1,
                       const unsigned short* __restrict__ O2T,
                       unsigned short* __restrict__ M) {
    __shared__ unsigned short Blds[2 * BBUF];
    int h    = blockIdx.x >> 5;            // 64 channels
    int tile = blockIdx.x & 31;            // 4 x 8 tiles of 128x64
    int ti = tile >> 3, tj = tile & 7;
    int wave = threadIdx.x >> 5;
    int lane = threadIdx.x & 31;
    int col  = lane & 15;
    int sel  = lane >> 4;
    int i0 = ti * 128 + wave * 16;
    int j0 = tj * 64;

    const unsigned short* Abase = O1 + (size_t)h * NPOS + (size_t)(i0 + col) * NN;
    // staging assignment: thread t copies 16B chunk (t&3) of row (t>>2)
    int srow = threadIdx.x >> 2, schk = threadIdx.x & 3;
    const unsigned short* Bglob = O2T + (size_t)h * NPOS + (size_t)(j0 + srow) * NN + schk * 8;
    unsigned short* Lws = &Blds[srow * BROW + schk * 8];

    // prologue: stage k-step 0 into buffer 0, load A(k0=0)
    ASYNC_CP16(Bglob, Lws);
    Frag Acur;
    Acur.q[0] = *(const uint4*)(Abase + sel * 8);
    Acur.q[1] = *(const uint4*)(Abase + 16 + sel * 8);

    v8f zro = {0.f,0.f,0.f,0.f,0.f,0.f,0.f,0.f};
    v8f acc[4] = {zro, zro, zro, zro};

    for (int s = 0; s < KSTEPS; ++s) {
        int k0  = s * 32;
        int cur = s & 1;
        bool more = (s + 1 < KSTEPS);
        if (more) ASYNC_CP16(Bglob + k0 + 32, Lws + ((s + 1) & 1) * BBUF);
        Frag Anext = Acur;
        if (more) {
            Anext.q[0] = *(const uint4*)(Abase + k0 + 32 + sel * 8);
            Anext.q[1] = *(const uint4*)(Abase + k0 + 48 + sel * 8);
        }
        if (more) { WAITA1(); } else { WAITA0(); }
        __syncthreads();                           // buffer `cur` published
        const unsigned short* Lbuf = &Blds[cur * BBUF];
        Frag Bf[4];
#pragma unroll
        for (int jt = 0; jt < 4; ++jt) {
            const unsigned short* bp = Lbuf + (jt * 16 + col) * BROW + sel * 8;
            Bf[jt].q[0] = *(const uint4*)bp;
            Bf[jt].q[1] = *(const uint4*)(bp + 16);
        }
#pragma unroll
        for (int jt = 0; jt < 4; ++jt)
            acc[jt] = __builtin_amdgcn_wmma_f32_16x16x32_bf16(
                false, Acur.v, false, Bf[jt].v, (short)0, acc[jt], false, false);
        __syncthreads();                           // done reading before overwrite
        Acur = Anext;
    }
#pragma unroll
    for (int jt = 0; jt < 4; ++jt) {
        int j = j0 + jt * 16 + col;
#pragma unroll
        for (int r = 0; r < 8; ++r) {
            int i = i0 + sel * 8 + r;
            M[((size_t)i * NN + j) * HH + h] = f2bf(acc[jt][r]);
        }
    }
}

// ----------------------------------------------------- batchnorm statistics
__global__ void k_bnstats(const float* __restrict__ V, float* ssum, float* ssq) {
    __shared__ float ls[256], lq[256];
    int c = threadIdx.x & 63, g = threadIdx.x >> 6;
    size_t base = (size_t)blockIdx.x * 1024;   // 256 blocks x 1024 positions
    float s = 0.f, q = 0.f;
    for (int pp = g; pp < 1024; pp += 4) {
        float v = V[(base + pp) * HH + c];
        s += v; q += v * v;
    }
    ls[threadIdx.x] = s; lq[threadIdx.x] = q;
    __syncthreads();
    if (g == 0) {
        for (int gg = 1; gg < 4; ++gg) { s += ls[gg * 64 + c]; q += lq[gg * 64 + c]; }
        atomicAdd(&ssum[c], s); atomicAdd(&ssq[c], q);
    }
}

__global__ void k_bnapply(const float* __restrict__ V, const float* __restrict__ ssum,
                          const float* __restrict__ ssq, const float* __restrict__ gam,
                          const float* __restrict__ bet, unsigned short* __restrict__ U) {
    size_t idx = (size_t)blockIdx.x * blockDim.x + threadIdx.x;
    int c = idx & 63;
    float mean = ssum[c] * (1.f / NPOS);
    float var  = ssq[c] * (1.f / NPOS) - mean * mean;
    float val  = (V[idx] - mean) * rsqrtf(var + BN_EPS) * gam[c] + bet[c];
    U[idx] = f2bf(val);
}

// ---------------------------------------------- extractor reductions + MLP
__global__ void k_exreduce(const unsigned short* __restrict__ U, int C,
                           float* total, float* trace) {
    __shared__ float ls[256];
    int c = threadIdx.x % C;
    int g = threadIdx.x / C;
    int groups = 256 / C;
    size_t base = (size_t)blockIdx.x * 1024;   // 256 blocks
    float s = 0.f;
    for (int pp = g; pp < 1024; pp += groups)
        s += bf2f(U[(base + pp) * C + c]);
    ls[threadIdx.x] = s;
    __syncthreads();
    if (g == 0) {
        for (int gg = 1; gg < groups; ++gg) s += ls[gg * C + c];
        atomicAdd(&total[c], s);
    }
    if (blockIdx.x == 0 && threadIdx.x < C) {
        float tr = 0.f;
        for (int i = 0; i < NN; ++i)
            tr += bf2f(U[((size_t)i * NN + i) * C + threadIdx.x]);
        trace[threadIdx.x] = tr;
    }
}

__global__ void k_exmlp(const float* __restrict__ trace, const float* __restrict__ total, int C,
                        const float* __restrict__ w1, const float* __restrict__ b1,
                        const float* __restrict__ w2, const float* __restrict__ w3,
                        float* outacc) {
    __shared__ float sh[HFD];
    int f = threadIdx.x;   // 64 threads
    float o = b1[f];
    for (int c = 0; c < C; ++c) {
        float t1 = trace[c] * (1.f / NN);
        float s  = (total[c] - trace[c]) * (1.f / ((float)NN * (NN - 1)));
        o += w1[f * C + c] * t1 + w2[f * C + c] * s;
    }
    sh[f] = o;
    __syncthreads();
    float o2 = o;
    for (int k = 0; k < HFD; ++k) o2 += w3[f * HFD + k] * fmaxf(sh[k], 0.f);
    outacc[f] += o2;
}

// ----------------------------------------------------------------- head
__global__ void k_head(const float* __restrict__ outacc,
                       const float* __restrict__ acw, const float* __restrict__ acb,
                       const float* __restrict__ flw, const float* __restrict__ flb,
                       float* __restrict__ dout) {
    __shared__ float sh[HFD], sh2[HFD], lg[NCLS];
    int f = threadIdx.x;
    if (f < HFD) sh[f] = fmaxf(outacc[f], 0.f) * (1.f / 3.f);
    __syncthreads();
    if (f < HFD) {
        float a = acb[f];
        for (int k = 0; k < HFD; ++k) a += acw[f * HFD + k] * sh[k];
        sh2[f] = sh[f] + fmaxf(a, 0.f);
    }
    __syncthreads();
    if (f < NCLS) {
        float l = flb[f];
        for (int k = 0; k < HFD; ++k) l += flw[f * HFD + k] * sh2[k];
        lg[f] = l;
    }
    __syncthreads();
    if (f < NCLS) {
        float m = lg[0];
        for (int k = 1; k < NCLS; ++k) m = fmaxf(m, lg[k]);
        float sum = 0.f;
        for (int k = 0; k < NCLS; ++k) sum += __expf(lg[k] - m);
        dout[f] = lg[f] - m - __logf(sum);
    }
}

// ========================================================================
extern "C" void kernel_launch(void* const* d_in, const int* in_sizes, int n_in,
                              void* d_out, int out_size, void* d_ws, size_t ws_size,
                              hipStream_t stream) {
    const float* x      = (const float*)d_in[0];
    const int*   ei     = (const int*)d_in[1];
    const float* np1_w  = (const float*)d_in[2];
    const float* np1_b  = (const float*)d_in[3];
    const float* np2_w  = (const float*)d_in[4];
    const float* np3_w  = (const float*)d_in[5];
    const float* c0m1_w = (const float*)d_in[6];
    const float* c0m1_b = (const float*)d_in[7];
    const float* c0m2_w = (const float*)d_in[8];
    const float* c0m2_b = (const float*)d_in[9];
    const float* c0m4_w = (const float*)d_in[10];
    const float* c0m4_b = (const float*)d_in[11];
    const float* cm1_w  = (const float*)d_in[12];
    const float* cm1_b  = (const float*)d_in[13];
    const float* cm2_w  = (const float*)d_in[14];
    const float* cm2_b  = (const float*)d_in[15];
    const float* cm4_w  = (const float*)d_in[16];
    const float* cm4_b  = (const float*)d_in[17];
    const float* bn_g   = (const float*)d_in[18];
    const float* bn_b   = (const float*)d_in[19];
    const float* fe1_w  = (const float*)d_in[20];
    const float* fe1_b  = (const float*)d_in[21];
    const float* fe2_w  = (const float*)d_in[22];
    const float* fe3_w  = (const float*)d_in[23];
    const float* ac_w   = (const float*)d_in[24];
    const float* ac_b   = (const float*)d_in[25];
    const float* fl_w   = (const float*)d_in[26];
    const float* fl_b   = (const float*)d_in[27];
    float* dout = (float*)d_out;

    char* ws = (char*)d_ws;
    unsigned short* U16 = (unsigned short*)(ws);                    // 32 MB u bf16 [p][C]
    unsigned short* O1  = (unsigned short*)(ws + (32ull << 20));    // 32 MB o1 [h][i][k]
    unsigned short* O2T = (unsigned short*)(ws + (64ull << 20));    // 32 MB o2^T [h][j][k]
    float*          V   = (float*)(ws + (32ull << 20));             // 64 MB (aliases O1+O2T)
    unsigned short* MUL = (unsigned short*)(ws + (96ull << 20));    // 32 MB mult [p][h]
    float*          AF  = (float*)(ws + (96ull << 20));             // 32 MB (aliases MUL)

    char* sm = ws + (128ull << 20);
    unsigned short* wb_c0m1 = (unsigned short*)(sm + 0);        // 64x32
    unsigned short* wb_c0m2 = (unsigned short*)(sm + 8192);     // 64x32
    unsigned short* wb_c0m4 = (unsigned short*)(sm + 16384);    // 64x96
    unsigned short* wb_cm1  = (unsigned short*)(sm + 32768);    // 2x64x64
    unsigned short* wb_cm2  = (unsigned short*)(sm + 65536);    // 2x64x64
    unsigned short* wb_cm4  = (unsigned short*)(sm + 98304);    // 2x64x128
    float* ssum   = (float*)(sm + 160 * 1024);
    float* ssq    = (float*)(sm + 160 * 1024 + 256);
    float* trace  = (float*)(sm + 160 * 1024 + 512);
    float* total  = (float*)(sm + 160 * 1024 + 768);
    float* outacc = (float*)(sm + 160 * 1024 + 1024);

    // --- prep: bf16 weights, zero A & accumulators
    k_cvt_bf16<<<8, 256, 0, stream>>>(c0m1_w, wb_c0m1, 64 * 32);
    k_cvt_bf16<<<8, 256, 0, stream>>>(c0m2_w, wb_c0m2, 64 * 32);
    k_cvt_bf16<<<24, 256, 0, stream>>>(c0m4_w, wb_c0m4, 64 * 96);
    k_cvt_bf16<<<32, 256, 0, stream>>>(cm1_w, wb_cm1, 2 * 64 * 64);
    k_cvt_bf16<<<32, 256, 0, stream>>>(cm2_w, wb_cm2, 2 * 64 * 64);
    k_cvt_bf16<<<64, 256, 0, stream>>>(cm4_w, wb_cm4, 2 * 64 * 128);
    k_zero_f32<<<NPOS * 32 / 256, 256, 0, stream>>>(AF, NPOS * 32);
    k_zero_f32<<<1, 64, 0, stream>>>(outacc, 64);

    // --- adjacency + u0
    k_build_A<<<NEDGE * 32 / 256, 256, 0, stream>>>(x, ei, AF);
    k_cvt_bf16<<<NPOS * 32 / 256, 256, 0, stream>>>(AF, U16, NPOS * 32);

    // --- no_prop extractor
    k_zero_f32<<<1, 64, 0, stream>>>(total, 64);
    k_exreduce<<<256, 256, 0, stream>>>(U16, 32, total, trace);
    k_exmlp<<<1, 64, 0, stream>>>(trace, total, 32, np1_w, np1_b, np2_w, np3_w, outacc);

    for (int l = 0; l < 3; ++l) {
        int Kin = (l == 0) ? 32 : 64;
        const unsigned short* w1 = (l == 0) ? wb_c0m1 : wb_cm1 + (size_t)(l - 1) * 4096;
        const unsigned short* w2 = (l == 0) ? wb_c0m2 : wb_cm2 + (size_t)(l - 1) * 4096;
        const unsigned short* w4 = (l == 0) ? wb_c0m4 : wb_cm4 + (size_t)(l - 1) * 8192;
        const float* b1 = (l == 0) ? c0m1_b : cm1_b + (l - 1) * 64;
        const float* b2 = (l == 0) ? c0m2_b : cm2_b + (l - 1) * 64;
        const float* b4 = (l == 0) ? c0m4_b : cm4_b + (l - 1) * 64;

        // o1 = 1x1conv(u) -> channel-major ; o2 -> channel-major transposed
        k_gemm_pos<<<2048, 256, 0, stream>>>(w1, b1, U16, Kin, nullptr, 0, O1, nullptr, 0);
        k_gemm_pos<<<2048, 256, 0, stream>>>(w2, b2, U16, Kin, nullptr, 0, O2T, nullptr, 1);
        // mult[h] = O1[h] @ O2[h]  (64 x 512^3 bf16 WMMA GEMM, async-LDS staged)
        k_mult<<<2048, 256, 0, stream>>>(O1, O2T, MUL);
        // u' = m4(concat(mult, u)) -> f32 position-major
        k_gemm_pos<<<2048, 256, 0, stream>>>(w4, b4, MUL, 64, U16, Kin, nullptr, V, 2);
        // batchnorm (training stats) -> u bf16
        k_zero_f32<<<1, 128, 0, stream>>>(ssum, 128);
        k_bnstats<<<256, 256, 0, stream>>>(V, ssum, ssq);
        k_bnapply<<<NPOS * 64 / 256, 256, 0, stream>>>(V, ssum, ssq,
                                                       bn_g + l * 64, bn_b + l * 64, U16);
        // per-layer extractor
        k_zero_f32<<<1, 64, 0, stream>>>(total, 64);
        k_exreduce<<<256, 256, 0, stream>>>(U16, 64, total, trace);
        k_exmlp<<<1, 64, 0, stream>>>(trace, total, 64,
                                      fe1_w + (size_t)l * 4096, fe1_b + l * 64,
                                      fe2_w + (size_t)l * 4096, fe3_w + (size_t)l * 4096,
                                      outacc);
    }

    k_head<<<1, 64, 0, stream>>>(outacc, ac_w, ac_b, fl_w, fl_b, dout);
}
